// ARMonocularModel_40046275068462
// MI455X (gfx1250) — matrix operand order repaired and checked by reading, not compiled, not run on hardware
//
#include <hip/hip_runtime.h>
#include <hip/hip_bf16.h>
#include <math.h>

// ---------------- model constants ----------------
#define D_MODEL 768
#define NHEAD   16
#define DHEAD   48
#define NLAYER  3
#define NTOK    256
#define BATCH   4
#define NPAST_  16
#define NFUT    20
#define CTX     (NTOK + 1 + NPAST_)   // 273
#define SMAX    (CTX + NFUT)          // 293
#define SPMAX   ((SMAX + 3) & ~3)     // 296 (padded score leading dim)

typedef __bf16 bf16;
typedef bf16  v16bf  __attribute__((ext_vector_type(16)));
typedef bf16  bf16x4 __attribute__((ext_vector_type(4)));
typedef float v8f    __attribute__((ext_vector_type(8)));
typedef float f32x4  __attribute__((ext_vector_type(4)));

// =====================================================================
// WMMA bf16 GEMM:  C = act( alpha * A(MxK) * B(KxN) + bias + residual )
// Batched over blockIdx.z with two-level stride (z / zdiv, z % zdiv).
// Block: 256 threads = 8 wave32, 32(M) x 64(N) tile, K-step 64 (2 WMMA).
// LDS tiles are double-buffered and stored in WMMA fragment order
// (ISA 7.12.2): each lane's 16 bf16 fragment elements are 32 contiguous
// bytes, so the fragment fetch is one v16bf load (2x ds_load_b128).
// Pipeline: stage(i+1) is issued before compute(i), one barrier per
// K-step. Bounds: M/N overruns only pollute never-stored outputs, so the
// row/col index is clamped (no zeroing); K overrun is zero-filled, and
// only the final K-step takes the guarded (clamp+select) staging path.
// =====================================================================
__global__ __launch_bounds__(256)
void gemm_bf16_wmma(const float* __restrict__ Abase, int lda, long sA, long sA2,
                    const float* __restrict__ Bbase, int ldb, long sB, long sB2, int transB,
                    const float* __restrict__ bias,
                    const float* __restrict__ resBase, long sR,
                    float* __restrict__ Cbase, int ldc, long sC, long sC2,
                    int M, int N, int K, float alpha, int act, int zdiv)
{
  const int z  = blockIdx.z;
  const int zb = z / zdiv, zh = z % zdiv;
  const float* A = Abase + (long)zb * sA + (long)zh * sA2;
  const float* B = Bbase + (long)zb * sB + (long)zh * sB2;
  const float* R = resBase ? (resBase + (long)zb * sR) : nullptr;
  float*       C = Cbase + (long)zb * sC + (long)zh * sC2;

  const int m0 = blockIdx.y * 32;
  const int n0 = blockIdx.x * 64;
  const int tid  = threadIdx.x;
  const int lane = tid & 31;
  const int wave = tid >> 5;
  const int wm = wave >> 2;   // 0..1  (M sub-tile)
  const int wn = wave & 3;    // 0..3  (N sub-tile)

  // [buffer][k-half][subtile][lane][elem] fragment-order tiles
  __shared__ __align__(32) bf16 AsF[2][2][2][32][16];
  __shared__ __align__(32) bf16 BsF[2][2][4][32][16];

  // Stage one 64-deep K-slab (32xA, 64xB) into buffer `buf`.
  auto stage = [&](int k0, int buf, bool guard) {
    // ---- A: 32 rows x 16 quads = 512 items, 2 per thread ----
#pragma unroll
    for (int w = 0; w < 2; ++w) {
      const int q  = tid + w * 256;
      const int r  = q >> 4;            // 0..31
      const int k  = (q & 15) * 4;      // 0..60
      const int gm = m0 + r;
      const long aRow = (long)((gm < M) ? gm : (M - 1)) * lda;  // clamp
      const int gk = k0 + k;
      f32x4 fv;
      if (!guard) {
        fv = *(const f32x4*)(A + aRow + gk);         // coalesced b128
        if (w == 0)                                  // speculative L2 pull
          __builtin_prefetch(A + aRow + gk + 64, 0, 1);
      } else {
#pragma unroll
        for (int j = 0; j < 4; ++j) {                // clamp + select
          int kc = gk + j; kc = (kc < K) ? kc : (K - 1);
          float t = A[aRow + kc];
          fv[j] = (gk + j < K) ? t : 0.f;
        }
      }
      bf16x4 bq;
#pragma unroll
      for (int j = 0; j < 4; ++j) bq[j] = (bf16)fv[j];
      const int kl = k & 31;
      *(bf16x4*)&AsF[buf][k >> 5][r >> 4]
                    [(r & 15) + ((kl & 8) ? 16 : 0)]
                    [(kl & 7) + ((kl & 16) ? 8 : 0)] = bq;
    }
    // ---- B: 64 cols x 16 quads = 1024 items, 4 per thread ----
#pragma unroll
    for (int w = 0; w < 4; ++w) {
      const int q = tid + w * 256;
      const int n = q & 63;
      const int k = (q >> 6) * 4;       // wave-uniform k quad
      int gn = n0 + n; gn = (gn < N) ? gn : (N - 1);  // clamp
      const int gk = k0 + k;
      f32x4 fv;
      if (transB) {
        const float* Brow = B + (long)gn * ldb;
        if (!guard) {
          fv = *(const f32x4*)(Brow + gk);
        } else {
#pragma unroll
          for (int j = 0; j < 4; ++j) {
            int kc = gk + j; kc = (kc < K) ? kc : (K - 1);
            float t = Brow[kc];
            fv[j] = (gk + j < K) ? t : 0.f;
          }
        }
      } else {
        if (!guard) {
#pragma unroll
          for (int j = 0; j < 4; ++j) fv[j] = B[(long)(gk + j) * ldb + gn];
        } else {
#pragma unroll
          for (int j = 0; j < 4; ++j) {
            int kc = gk + j; kc = (kc < K) ? kc : (K - 1);
            float t = B[(long)kc * ldb + gn];
            fv[j] = (gk + j < K) ? t : 0.f;
          }
        }
      }
      bf16x4 bq;
#pragma unroll
      for (int j = 0; j < 4; ++j) bq[j] = (bf16)fv[j];
      const int kl = k & 31;
      *(bf16x4*)&BsF[buf][k >> 5][n >> 4]
                    [(n & 15) + ((kl & 8) ? 16 : 0)]
                    [(kl & 7) + ((kl & 16) ? 8 : 0)] = bq;
    }
  };

  v8f acc = {0.f,0.f,0.f,0.f,0.f,0.f,0.f,0.f};

  const int  nIter = (K + 63) >> 6;
  const bool tailG = (K & 63) != 0;

  stage(0, 0, (nIter == 1) && tailG);

  for (int i = 0; i < nIter; ++i) {
    __syncthreads();                       // buffer i&1 is ready
    if (i + 1 < nIter)                     // overlap next slab with WMMAs
      stage((i + 1) * 64, (i + 1) & 1, (i + 1 == nIter - 1) && tailG);

    const int buf = i & 1;
    v16bf a0 = *(const v16bf*)AsF[buf][0][wm][lane];
    v16bf b0 = *(const v16bf*)BsF[buf][0][wn][lane];
    v16bf a1 = *(const v16bf*)AsF[buf][1][wm][lane];
    v16bf b1 = *(const v16bf*)BsF[buf][1][wn][lane];
    acc = __builtin_amdgcn_wmma_f32_16x16x32_bf16(
        false, a0, false, b0, (short)0, acc, false, false);
    acc = __builtin_amdgcn_wmma_f32_16x16x32_bf16(
        false, a1, false, b1, (short)0, acc, false, false);
  }

  // ---- epilogue: D layout VGPR r -> M = r + (lane>=16 ? 8 : 0), N = lane&15
  const int gm0 = m0 + wm * 16 + ((lane & 16) ? 8 : 0);
  const int gn  = n0 + wn * 16 + (lane & 15);
#pragma unroll
  for (int r = 0; r < 8; ++r) {
    int gm = gm0 + r;
    if (gm < M && gn < N) {
      float v = acc[r] * alpha;
      if (bias) v += bias[gn];
      if (R)    v += R[(long)gm * ldc + gn];
      if (act == 1) v = fmaxf(v, 0.f);
      C[(long)gm * ldc + gn] = v;
    }
  }
}

// =====================================================================
// LayerNorm over D_MODEL, one block per row.
// =====================================================================
__global__ __launch_bounds__(256)
void layernorm_kernel(const float* __restrict__ x, float* __restrict__ y,
                      const float* __restrict__ g, const float* __restrict__ b)
{
  const long row = blockIdx.x;
  const float* xr = x + row * D_MODEL;
  float*       yr = y + row * D_MODEL;
  __shared__ float red[256];

  float s = 0.f;
  for (int i = threadIdx.x; i < D_MODEL; i += 256) s += xr[i];
  red[threadIdx.x] = s; __syncthreads();
  for (int st = 128; st > 0; st >>= 1) {
    if (threadIdx.x < st) red[threadIdx.x] += red[threadIdx.x + st];
    __syncthreads();
  }
  const float mean = red[0] / D_MODEL;
  __syncthreads();

  float s2 = 0.f;
  for (int i = threadIdx.x; i < D_MODEL; i += 256) { float d = xr[i] - mean; s2 += d * d; }
  red[threadIdx.x] = s2; __syncthreads();
  for (int st = 128; st > 0; st >>= 1) {
    if (threadIdx.x < st) red[threadIdx.x] += red[threadIdx.x + st];
    __syncthreads();
  }
  const float inv = rsqrtf(red[0] / D_MODEL + 1e-5f);
  for (int i = threadIdx.x; i < D_MODEL; i += 256)
    yr[i] = (xr[i] - mean) * inv * g[i] + b[i];
}

// =====================================================================
// Masked softmax in place (leading dim Sp >= S, multiple of 4).
// Row qi >= CTX attends causally (cols <= qi); masked probs written as 0
// so the attn*V GEMM runs full-K.   grid = (S, B*H)
// =====================================================================
__global__ __launch_bounds__(256)
void softmax_mask_kernel(float* __restrict__ scores, int S, int Sp)
{
  const int qi = blockIdx.x;
  float* row = scores + (long)blockIdx.y * Sp * Sp + (long)qi * Sp;
  const int limit = (qi >= CTX) ? (qi + 1) : S;
  __shared__ float red[256];

  float mx = -1e30f;
  for (int j = threadIdx.x; j < limit; j += 256) mx = fmaxf(mx, row[j]);
  red[threadIdx.x] = mx; __syncthreads();
  for (int st = 128; st > 0; st >>= 1) {
    if (threadIdx.x < st) red[threadIdx.x] = fmaxf(red[threadIdx.x], red[threadIdx.x + st]);
    __syncthreads();
  }
  mx = red[0]; __syncthreads();

  float sum = 0.f;
  for (int j = threadIdx.x; j < limit; j += 256) {
    float e = __expf(row[j] - mx); row[j] = e; sum += e;
  }
  red[threadIdx.x] = sum; __syncthreads();
  for (int st = 128; st > 0; st >>= 1) {
    if (threadIdx.x < st) red[threadIdx.x] += red[threadIdx.x + st];
    __syncthreads();
  }
  const float inv = 1.f / red[0];
  for (int j = threadIdx.x; j < S; j += 256)
    row[j] = (j < limit) ? row[j] * inv : 0.f;
}

// =====================================================================
// Context assembly: [image+pos | intent token | past projection + time emb]
// =====================================================================
__global__ void context_prep_kernel(const float* __restrict__ img, const float* __restrict__ pos,
                                    const int* __restrict__ intent, const float* __restrict__ iemb,
                                    const float* __restrict__ past,
                                    const float* __restrict__ W_past, const float* __restrict__ b_past,
                                    const float* __restrict__ W_ppos, const float* __restrict__ b_ppos,
                                    const float* __restrict__ temb, float* __restrict__ ctx)
{
  long idx = (long)blockIdx.x * blockDim.x + threadIdx.x;
  const long total = (long)BATCH * CTX * D_MODEL;
  if (idx >= total) return;
  const int d = (int)(idx % D_MODEL);
  const long rb = idx / D_MODEL;
  const int row = (int)(rb % CTX);
  const int b   = (int)(rb / CTX);
  float v;
  if (row < NTOK) {
    v = img[((long)b * NTOK + row) * D_MODEL + d] + pos[(long)row * D_MODEL + d];
  } else if (row == NTOK) {
    int ii = intent[b] - 1; ii = ii < 0 ? 0 : (ii > 2 ? 2 : ii);
    v = iemb[(long)ii * D_MODEL + d];
  } else {
    const int j = row - NTOK - 1;
    const float* p = past + ((long)b * NPAST_ + j) * 6;
    float s = b_past[d] + b_ppos[d] + temb[(long)j * D_MODEL + d];
#pragma unroll
    for (int i = 0; i < 6; ++i) s += p[i] * W_past[(long)i * D_MODEL + d];
    s += p[0] * W_ppos[d] + p[1] * W_ppos[D_MODEL + d];
    v = s;
  }
  ctx[idx] = v;
}

__global__ void future_init_kernel(const float* __restrict__ fq, const float* __restrict__ temb,
                                   float* __restrict__ future)
{
  long idx = (long)blockIdx.x * blockDim.x + threadIdx.x;
  const long total = (long)BATCH * NFUT * D_MODEL;
  if (idx >= total) return;
  const int d = (int)(idx % D_MODEL);
  const int f = (int)((idx / D_MODEL) % NFUT);
  future[idx] = fq[(long)f * D_MODEL + d] + temb[(long)(NPAST_ + f) * D_MODEL + d];
}

__global__ void assemble_x_kernel(const float* __restrict__ ctx, const float* __restrict__ future,
                                  float* __restrict__ x, int S)
{
  long idx = (long)blockIdx.x * blockDim.x + threadIdx.x;
  const long total = (long)BATCH * S * D_MODEL;
  if (idx >= total) return;
  const int d = (int)(idx % D_MODEL);
  const long rb = idx / D_MODEL;
  const int row = (int)(rb % S);
  const int b   = (int)(rb / S);
  x[idx] = (row < CTX) ? ctx[((long)b * CTX + row) * D_MODEL + d]
                       : future[((long)b * NFUT + (row - CTX)) * D_MODEL + d];
}

// =====================================================================
// Tiny decoder GEMM (M=4): one thread per output, exact GELU option.
// =====================================================================
__global__ void small_mm_kernel(const float* __restrict__ A, long sA,
                                const float* __restrict__ W, const float* __restrict__ bias,
                                float* __restrict__ C, int M, int N, int K, int act)
{
  int idx = blockIdx.x * blockDim.x + threadIdx.x;
  if (idx >= M * N) return;
  const int m = idx / N, n = idx % N;
  const float* a = A + (long)m * sA;
  float s = bias ? bias[n] : 0.f;
  for (int k = 0; k < K; ++k) s += a[k] * W[(long)k * N + n];
  if (act == 1) s = 0.5f * s * (1.f + erff(s * 0.70710678118654752f));
  C[idx] = s;
}

__global__ void store_pred_kernel(const float* __restrict__ p, float* __restrict__ out, int t)
{
  int i = threadIdx.x;
  if (i < BATCH * 2) {
    int b = i >> 1, c = i & 1;
    out[((long)b * NFUT + t) * 2 + c] = p[i];
  }
}

// future[:, t+1] += relu( LN( p @ W_pp + b_pp ) * g + be )
__global__ __launch_bounds__(256)
void future_update_kernel(const float* __restrict__ p,
                          const float* __restrict__ W_pp, const float* __restrict__ b_pp,
                          const float* __restrict__ g_pp, const float* __restrict__ be_pp,
                          float* __restrict__ future, int t)
{
  const int b = blockIdx.x;
  const float p0 = p[b * 2 + 0], p1 = p[b * 2 + 1];
  __shared__ float u[D_MODEL];
  __shared__ float red[256];

  float s = 0.f;
  for (int i = threadIdx.x; i < D_MODEL; i += 256) {
    float v = p0 * W_pp[i] + p1 * W_pp[D_MODEL + i] + b_pp[i];
    u[i] = v; s += v;
  }
  red[threadIdx.x] = s; __syncthreads();
  for (int st = 128; st > 0; st >>= 1) {
    if (threadIdx.x < st) red[threadIdx.x] += red[threadIdx.x + st];
    __syncthreads();
  }
  const float mean = red[0] / D_MODEL; __syncthreads();
  float s2 = 0.f;
  for (int i = threadIdx.x; i < D_MODEL; i += 256) { float d = u[i] - mean; s2 += d * d; }
  red[threadIdx.x] = s2; __syncthreads();
  for (int st = 128; st > 0; st >>= 1) {
    if (threadIdx.x < st) red[threadIdx.x] += red[threadIdx.x + st];
    __syncthreads();
  }
  const float inv = rsqrtf(red[0] / D_MODEL + 1e-5f);
  float* f = future + ((long)b * NFUT + (t + 1)) * D_MODEL;
  for (int i = threadIdx.x; i < D_MODEL; i += 256) {
    float v = (u[i] - mean) * inv * g_pp[i] + be_pp[i];
    f[i] += fmaxf(v, 0.f);
  }
}

// =====================================================================
// Host orchestration
// =====================================================================
extern "C" void kernel_launch(void* const* d_in, const int* in_sizes, int n_in,
                              void* d_out, int out_size, void* d_ws, size_t ws_size,
                              hipStream_t stream)
{
  (void)in_sizes; (void)n_in; (void)out_size; (void)ws_size;
  const int Dm = D_MODEL;

  const float* img    = (const float*)d_in[0];
  const float* past   = (const float*)d_in[1];
  const int*   intent = (const int*)  d_in[2];
  const float* pos    = (const float*)d_in[3];
  const float* fq     = (const float*)d_in[4];
  const float* iemb   = (const float*)d_in[5];
  const float* temb   = (const float*)d_in[6];
  const float* W_past = (const float*)d_in[7];
  const float* b_past = (const float*)d_in[8];
  const float* W_ppos = (const float*)d_in[9];
  const float* b_ppos = (const float*)d_in[10];
  const float* W_pp   = (const float*)d_in[11];
  const float* b_pp   = (const float*)d_in[12];
  const float* g_pp   = (const float*)d_in[13];
  const float* be_pp  = (const float*)d_in[14];
  const float* Wqkv   = (const float*)d_in[15];
  const float* bqkv   = (const float*)d_in[16];
  const float* Wo     = (const float*)d_in[17];
  const float* bo     = (const float*)d_in[18];
  const float* g1     = (const float*)d_in[19];
  const float* beta1  = (const float*)d_in[20];
  const float* g2     = (const float*)d_in[21];
  const float* beta2  = (const float*)d_in[22];
  const float* W1     = (const float*)d_in[23];
  const float* bf1    = (const float*)d_in[24];
  const float* W2     = (const float*)d_in[25];
  const float* bf2    = (const float*)d_in[26];
  const float* Wd1    = (const float*)d_in[27];
  const float* bd1    = (const float*)d_in[28];
  const float* Wd2    = (const float*)d_in[29];
  const float* bd2    = (const float*)d_in[30];
  const float* Wd3    = (const float*)d_in[31];
  const float* bd3    = (const float*)d_in[32];
  float* out = (float*)d_out;

  // ---- workspace carve (floats) ----
  float* ws = (float*)d_ws;
  float* ctx    = ws; ws += (long)BATCH * CTX  * Dm;
  float* future = ws; ws += (long)BATCH * NFUT * Dm;
  float* x      = ws; ws += (long)BATCH * SMAX * Dm;
  float* h      = ws; ws += (long)BATCH * SMAX * Dm;
  float* qkv    = ws; ws += (long)BATCH * SMAX * 3 * Dm;
  float* attn   = ws; ws += (long)BATCH * SMAX * Dm;
  float* ffn    = ws; ws += (long)BATCH * SMAX * 4 * Dm;
  float* scores = ws; ws += (long)BATCH * NHEAD * SPMAX * SPMAX;
  float* dec1   = ws; ws += (long)BATCH * Dm;
  float* dec2   = ws; ws += (long)BATCH * Dm;
  float* pbuf   = ws; ws += (long)BATCH * 2;

  {
    long tot = (long)BATCH * CTX * Dm;
    context_prep_kernel<<<dim3((unsigned)((tot + 255) / 256)), 256, 0, stream>>>(
        img, pos, intent, iemb, past, W_past, b_past, W_ppos, b_ppos, temb, ctx);
    long tf = (long)BATCH * NFUT * Dm;
    future_init_kernel<<<dim3((unsigned)((tf + 255) / 256)), 256, 0, stream>>>(fq, temb, future);
  }

  const float inv_sqrt_dh = 1.0f / sqrtf((float)DHEAD);

  for (int t = 0; t < NFUT; ++t) {
    const int S  = CTX + t + 1;
    const int Sp = (S + 3) & ~3;   // padded score leading dim (16B-aligned rows)
    {
      long tot = (long)BATCH * S * Dm;
      assemble_x_kernel<<<dim3((unsigned)((tot + 255) / 256)), 256, 0, stream>>>(ctx, future, x, S);
    }

    for (int l = 0; l < NLAYER; ++l) {
      const float* Wqkv_l = Wqkv + (long)l * Dm * 3 * Dm;
      const float* bqkv_l = bqkv + (long)l * 3 * Dm;
      const float* Wo_l   = Wo   + (long)l * Dm * Dm;
      const float* bo_l   = bo   + (long)l * Dm;
      const float* W1_l   = W1   + (long)l * Dm * 4 * Dm;
      const float* bf1_l  = bf1  + (long)l * 4 * Dm;
      const float* W2_l   = W2   + (long)l * 4 * Dm * Dm;
      const float* bf2_l  = bf2  + (long)l * Dm;

      // h = LN1(x)
      layernorm_kernel<<<dim3(BATCH * S), 256, 0, stream>>>(x, h, g1 + l * Dm, beta1 + l * Dm);

      // qkv = h @ Wqkv + bqkv
      gemm_bf16_wmma<<<dim3(3 * Dm / 64, (S + 31) / 32, BATCH), 256, 0, stream>>>(
          h, Dm, (long)S * Dm, 0,
          Wqkv_l, 3 * Dm, 0, 0, 0,
          bqkv_l, nullptr, 0,
          qkv, 3 * Dm, (long)S * 3 * Dm, 0,
          S, 3 * Dm, Dm, 1.f, 0, 1);

      // scores = q @ k^T / sqrt(DH)   (per batch*head; ldc = Sp)
      gemm_bf16_wmma<<<dim3((S + 63) / 64, (S + 31) / 32, BATCH * NHEAD), 256, 0, stream>>>(
          qkv, 3 * Dm, (long)S * 3 * Dm, DHEAD,
          qkv + Dm, 3 * Dm, (long)S * 3 * Dm, DHEAD, 1,
          nullptr, nullptr, 0,
          scores, Sp, (long)NHEAD * Sp * Sp, (long)Sp * Sp,
          S, S, DHEAD, inv_sqrt_dh, 0, NHEAD);

      softmax_mask_kernel<<<dim3(S, BATCH * NHEAD), 256, 0, stream>>>(scores, S, Sp);

      // attn = probs @ v   (per batch*head, N = 48, lda = Sp)
      gemm_bf16_wmma<<<dim3(1, (S + 31) / 32, BATCH * NHEAD), 256, 0, stream>>>(
          scores, Sp, (long)NHEAD * Sp * Sp, (long)Sp * Sp,
          qkv + 2 * Dm, 3 * Dm, (long)S * 3 * Dm, DHEAD, 0,
          nullptr, nullptr, 0,
          attn, Dm, (long)S * Dm, DHEAD,
          S, DHEAD, S, 1.f, 0, NHEAD);

      // x = x + attn @ Wo + bo
      gemm_bf16_wmma<<<dim3(Dm / 64, (S + 31) / 32, BATCH), 256, 0, stream>>>(
          attn, Dm, (long)S * Dm, 0,
          Wo_l, Dm, 0, 0, 0,
          bo_l, x, (long)S * Dm,
          x, Dm, (long)S * Dm, 0,
          S, Dm, Dm, 1.f, 0, 1);

      // h = LN2(x)
      layernorm_kernel<<<dim3(BATCH * S), 256, 0, stream>>>(x, h, g2 + l * Dm, beta2 + l * Dm);

      // ffn = relu(h @ W1 + bf1)
      gemm_bf16_wmma<<<dim3(4 * Dm / 64, (S + 31) / 32, BATCH), 256, 0, stream>>>(
          h, Dm, (long)S * Dm, 0,
          W1_l, 4 * Dm, 0, 0, 0,
          bf1_l, nullptr, 0,
          ffn, 4 * Dm, (long)S * 4 * Dm, 0,
          S, 4 * Dm, Dm, 1.f, 1, 1);

      // x = x + ffn @ W2 + bf2
      gemm_bf16_wmma<<<dim3(Dm / 64, (S + 31) / 32, BATCH), 256, 0, stream>>>(
          ffn, 4 * Dm, (long)S * 4 * Dm, 0,
          W2_l, Dm, 0, 0, 0,
          bf2_l, x, (long)S * Dm,
          x, Dm, (long)S * Dm, 0,
          S, Dm, 4 * Dm, 1.f, 0, 1);
    }

    // decoder head on last token
    small_mm_kernel<<<dim3((BATCH * Dm + 255) / 256), 256, 0, stream>>>(
        x + (long)(S - 1) * Dm, (long)S * Dm, Wd1, bd1, dec1, BATCH, Dm, Dm, 1);
    small_mm_kernel<<<dim3((BATCH * Dm + 255) / 256), 256, 0, stream>>>(
        dec1, Dm, Wd2, bd2, dec2, BATCH, Dm, Dm, 1);
    small_mm_kernel<<<dim3(1), 256, 0, stream>>>(
        dec2, Dm, Wd3, bd3, pbuf, BATCH, 2, Dm, 0);

    store_pred_kernel<<<dim3(1), 32, 0, stream>>>(pbuf, out, t);

    if (t < NFUT - 1) {
      future_update_kernel<<<dim3(BATCH), 256, 0, stream>>>(
          pbuf, W_pp, b_pp, g_pp, be_pp, future, t);
    }
  }
}
